// DiscreteKeyValueBottleneck_37907381355000
// MI455X (gfx1250) — matrix operand order repaired and checked by reading, not compiled, not run on hardware
//
#include <hip/hip_runtime.h>

// ---------------------------------------------------------------------------
// Discrete key-value bottleneck (VQ argmin + value gather) for gfx1250.
//   dots  = x @ codebook^T        (M=32768, N=8192, K=512)
//   idx   = argmin_m (||c_m||^2 - 2*dots)
//   out   = values[idx]
// Split-bf16 (hi/lo) WMMA GEMM fused with running argmin:
//   dot_f32 ~= Ahi*Bhi + Ahi*Blo + Alo*Bhi  -> 3 independent WMMA acc chains.
// Codebook tiles are streamed into LDS with global_load_async_to_lds_b128
// (ASYNCcnt) using half-K double buffering so staging overlaps compute.
// ---------------------------------------------------------------------------

typedef __bf16 bf16_t;
typedef bf16_t v16bf __attribute__((ext_vector_type(16)));
typedef bf16_t v8bf  __attribute__((ext_vector_type(8)));
typedef bf16_t v4bf  __attribute__((ext_vector_type(4)));
typedef float  v8f   __attribute__((ext_vector_type(8)));

#define DIM        512
#define NMEM       8192
#define MROWS      (8 * 4096)
#define BLOCK_ROWS 128          // 8 waves x 16 rows
#define NTHREADS   256
#define LDSK       520          // x-tile row stride (elements), bank friendly
#define KHALF      256          // K per pipeline phase
#define CBK        264          // cb half-tile row stride (elements)
#define NPHASE     (2 * (NMEM / 16))

// -------------------------------------------------------------------------
// Prep: codebook f32 -> (hi, lo) bf16 split + ||c||^2
// -------------------------------------------------------------------------
__global__ __launch_bounds__(256) void vq_prep(const float* __restrict__ cb,
                                               bf16_t* __restrict__ cb_hi,
                                               bf16_t* __restrict__ cb_lo,
                                               float* __restrict__ c_sq) {
    const int row  = blockIdx.x;
    const int t    = threadIdx.x;
    const int lane = t & 31;
    const int wave = t >> 5;

    float s = 0.0f;
#pragma unroll
    for (int i = 0; i < 2; ++i) {
        const int c = t + i * 256;
        const float f = cb[(size_t)row * DIM + c];
        const bf16_t hi = (bf16_t)f;
        const bf16_t lo = (bf16_t)(f - (float)hi);
        cb_hi[(size_t)row * DIM + c] = hi;
        cb_lo[(size_t)row * DIM + c] = lo;
        s += f * f;
    }
#pragma unroll
    for (int off = 16; off > 0; off >>= 1) s += __shfl_down(s, off, 32);

    __shared__ float red[8];
    if (lane == 0) red[wave] = s;
    __syncthreads();
    if (t == 0) {
        float tot = 0.0f;
#pragma unroll
        for (int w = 0; w < 8; ++w) tot += red[w];
        c_sq[row] = tot;
    }
}

// -------------------------------------------------------------------------
// Main fused kernel
// -------------------------------------------------------------------------
__global__ __launch_bounds__(NTHREADS) void vq_main(const float* __restrict__ x,
                                                    const bf16_t* __restrict__ cb_hi,
                                                    const bf16_t* __restrict__ cb_lo,
                                                    const float* __restrict__ c_sq,
                                                    const float* __restrict__ values,
                                                    float* __restrict__ out) {
    extern __shared__ char smem_raw[];
    bf16_t* xhi = (bf16_t*)smem_raw;                 // [BLOCK_ROWS][LDSK]
    bf16_t* xlo = xhi + BLOCK_ROWS * LDSK;
    bf16_t* cbh[2];
    bf16_t* cbl[2];
    cbh[0] = xlo + BLOCK_ROWS * LDSK;                // [16][CBK] half-K tiles
    cbl[0] = cbh[0] + 16 * CBK;
    cbh[1] = cbl[0] + 16 * CBK;
    cbl[1] = cbh[1] + 16 * CBK;

    const int t    = threadIdx.x;
    const int lane = t & 31;
    const int wave = t >> 5;
    const int row0 = blockIdx.x * BLOCK_ROWS;

    // ---- stage x tile: f32 -> hi/lo bf16 into LDS -------------------------
    const float4* x4 = (const float4*)x;
#pragma unroll 4
    for (int i = 0; i < 64; ++i) {
        const int f4 = t + i * 256;                  // 0..16383 float4 chunks
        const int r  = f4 >> 7;                      // row within tile
        const int c  = (f4 & 127) << 2;              // col (elements)
        const float4 v = x4[(size_t)(row0 + r) * (DIM / 4) + (c >> 2)];
        v4bf h, l;
        const float fs[4] = {v.x, v.y, v.z, v.w};
#pragma unroll
        for (int j = 0; j < 4; ++j) {
            const bf16_t hj = (bf16_t)fs[j];
            h[j] = hj;
            l[j] = (bf16_t)(fs[j] - (float)hj);
        }
        *(v4bf*)&xhi[r * LDSK + c] = h;
        *(v4bf*)&xlo[r * LDSK + c] = l;
    }

    // ---- per-lane WMMA fragment addressing (ISA layout tables) ------------
    const int half = lane >> 4;          // 0: lanes 0-15, 1: lanes 16-31
    const int l16  = lane & 15;
    const int akb  = half * 8;           // A: K-base per lane half
    const int bkb  = half * 8;           // B: K-base per lane half
    const bf16_t* aRowH = xhi + (wave * 16 + l16) * LDSK;
    const bf16_t* aRowL = xlo + (wave * 16 + l16) * LDSK;

    float best_d[8];
    int   best_i[8];
#pragma unroll
    for (int r = 0; r < 8; ++r) { best_d[r] = __builtin_inff(); best_i[r] = 0; }

    // ---- async staging of one half-K codebook tile ------------------------
    // Per thread: 2 chunks x 16B of hi + same of lo -> 4 async b128 loads.
    auto stage = [&](int p) {
        const int n0 = (p >> 1) * 16;
        const int kh = (p & 1) * KHALF;
        bf16_t* dH = cbh[p & 1];
        bf16_t* dL = cbl[p & 1];
#pragma unroll
        for (int i = 0; i < 2; ++i) {
            const int ch = t + i * 256;              // 0..511 16B chunks
            const int r  = ch >> 5;                  // code row 0..15
            const int c  = (ch & 31) << 3;           // col 0..248 step 8
            const unsigned int ldsH = (unsigned int)(uintptr_t)(dH + r * CBK + c);
            const unsigned int ldsL = (unsigned int)(uintptr_t)(dL + r * CBK + c);
            const unsigned int voff = (unsigned int)(((n0 + r) * DIM + kh + c) * 2);
            asm volatile("global_load_async_to_lds_b128 %0, %1, %2"
                         :: "v"(ldsH), "v"(voff), "s"(cb_hi) : "memory");
            asm volatile("global_load_async_to_lds_b128 %0, %1, %2"
                         :: "v"(ldsL), "v"(voff), "s"(cb_lo) : "memory");
        }
    };

    stage(0);   // prologue: phase 0 in flight

    // ---- scan all codebook tiles (2 pipeline phases per 16-code tile) -----
    for (int nt = 0; nt < NMEM / 16; ++nt) {
        v8f ahh = {0.f, 0.f, 0.f, 0.f, 0.f, 0.f, 0.f, 0.f};
        v8f ahl = ahh;
        v8f alh = ahh;

#pragma unroll
        for (int h = 0; h < 2; ++h) {
            const int p = nt * 2 + h;
            asm volatile("s_wait_asynccnt 0x0" ::: "memory");   // my phase-p data in LDS
            __syncthreads();                                     // everyone's data + buf free
            if (p + 1 < NPHASE) stage(p + 1);                    // overlap next phase

            const bf16_t* aH = aRowH + h * KHALF + akb;
            const bf16_t* aL = aRowL + h * KHALF + akb;
            const bf16_t* bH = cbh[h] + l16 * CBK + bkb;
            const bf16_t* bL = cbl[h] + l16 * CBK + bkb;
#pragma unroll
            for (int ks = 0; ks < 8; ++ks) {
                const int k0 = ks * 32;
                // A frag: elements [k0+akb, +8) and [k0+akb+16, +8)
                v8bf ah0 = *(const v8bf*)(aH + k0);
                v8bf ah1 = *(const v8bf*)(aH + k0 + 16);
                v8bf al0 = *(const v8bf*)(aL + k0);
                v8bf al1 = *(const v8bf*)(aL + k0 + 16);
                // B frag: elements [k0+bkb, +8) and [k0+bkb+16, +8)
                v8bf bh0 = *(const v8bf*)(bH + k0);
                v8bf bh1 = *(const v8bf*)(bH + k0 + 16);
                v8bf bl0 = *(const v8bf*)(bL + k0);
                v8bf bl1 = *(const v8bf*)(bL + k0 + 16);
                v16bf AH = __builtin_shufflevector(ah0, ah1, 0,1,2,3,4,5,6,7,8,9,10,11,12,13,14,15);
                v16bf AL = __builtin_shufflevector(al0, al1, 0,1,2,3,4,5,6,7,8,9,10,11,12,13,14,15);
                v16bf BH = __builtin_shufflevector(bh0, bh1, 0,1,2,3,4,5,6,7,8,9,10,11,12,13,14,15);
                v16bf BL = __builtin_shufflevector(bl0, bl1, 0,1,2,3,4,5,6,7,8,9,10,11,12,13,14,15);
                // three independent accumulation chains
                ahh = __builtin_amdgcn_wmma_f32_16x16x32_bf16(false, AH, false, BH, (short)0, ahh, false, false);
                ahl = __builtin_amdgcn_wmma_f32_16x16x32_bf16(false, AH, false, BL, (short)0, ahl, false, false);
                alh = __builtin_amdgcn_wmma_f32_16x16x32_bf16(false, AL, false, BH, (short)0, alh, false, false);
            }
        }

        // distance + running argmin (lane holds rows 8*half+r, code n0+l16)
        const int   n0   = nt * 16;
        const float cs   = c_sq[n0 + l16];
        const int   code = n0 + l16;
#pragma unroll
        for (int r = 0; r < 8; ++r) {
            const float dot = ahh[r] + (ahl[r] + alh[r]);
            const float d   = fmaf(-2.0f, dot, cs);
            if (d < best_d[r] || (d == best_d[r] && code < best_i[r])) {
                best_d[r] = d;
                best_i[r] = code;
            }
        }
    }

    // ---- cross-lane argmin reduction within each 16-lane half -------------
#pragma unroll
    for (int st = 1; st <= 8; st <<= 1) {
#pragma unroll
        for (int r = 0; r < 8; ++r) {
            const float od = __shfl_xor(best_d[r], st, 32);
            const int   oi = __shfl_xor(best_i[r], st, 32);
            if (od < best_d[r] || (od == best_d[r] && oi < best_i[r])) {
                best_d[r] = od;
                best_i[r] = oi;
            }
        }
    }

    // ---- gather values rows (wave-cooperative float4 copy) ----------------
    const float4* vals4 = (const float4*)values;
    float4*       out4  = (float4*)out;
#pragma unroll
    for (int rr = 0; rr < 16; ++rr) {
        const int idx = __shfl(best_i[rr & 7], (rr >> 3) * 16, 32);
        const size_t srow = (size_t)idx * (DIM / 4);
        const size_t drow = (size_t)(row0 + wave * 16 + rr) * (DIM / 4);
#pragma unroll
        for (int j = 0; j < 4; ++j)
            out4[drow + lane + j * 32] = vals4[srow + lane + j * 32];
    }
}

// -------------------------------------------------------------------------
extern "C" void kernel_launch(void* const* d_in, const int* in_sizes, int n_in,
                              void* d_out, int out_size, void* d_ws, size_t ws_size,
                              hipStream_t stream) {
    (void)in_sizes; (void)n_in; (void)out_size; (void)ws_size;
    const float* x        = (const float*)d_in[0];
    const float* codebook = (const float*)d_in[1];
    const float* values   = (const float*)d_in[2];
    float*       out      = (float*)d_out;

    // workspace layout: cb_hi (8MB) | cb_lo (8MB) | c_sq (32KB)
    bf16_t* cb_hi = (bf16_t*)d_ws;
    bf16_t* cb_lo = cb_hi + (size_t)NMEM * DIM;
    float*  csq   = (float*)(cb_lo + (size_t)NMEM * DIM);

    vq_prep<<<NMEM, 256, 0, stream>>>(codebook, cb_hi, cb_lo, csq);

    constexpr size_t smem =
        (size_t)(2 * BLOCK_ROWS * LDSK + 4 * 16 * CBK) * sizeof(bf16_t);
    static_assert(smem <= 320u * 1024u, "LDS over budget");
    hipFuncSetAttribute((const void*)vq_main,
                        hipFuncAttributeMaxDynamicSharedMemorySize, (int)smem);
    vq_main<<<MROWS / BLOCK_ROWS, NTHREADS, smem, stream>>>(x, cb_hi, cb_lo, csq,
                                                            values, out);
}
